// CrossAttention_14680198218157
// MI455X (gfx1250) — matrix operand order repaired and checked by reading, not compiled, not run on hardware
//
#include <hip/hip_runtime.h>

typedef __attribute__((ext_vector_type(16))) __bf16 v16bf;
typedef __attribute__((ext_vector_type(8)))  __bf16 v8bf;
typedef __attribute__((ext_vector_type(8)))  float  v8f;
typedef __attribute__((ext_vector_type(4)))  float  v4f;

#define WMMA_BF16(a, b, c) \
  __builtin_amdgcn_wmma_f32_16x16x32_bf16(false, (a), false, (b), (short)0, (c), false, false)

// Problem constants: B=8, L=2048, DQ=512, DKV=768, H=512

// ---------------------------------------------------------------------------
// Prep kernels: fp32 -> bf16 conversion, and transpose+convert for weights
// ---------------------------------------------------------------------------
__global__ void cvt_f32_bf16(const float* __restrict__ src, __bf16* __restrict__ dst, int n) {
  int i = blockIdx.x * blockDim.x + threadIdx.x;
  int stride = gridDim.x * blockDim.x;
  for (; i < n; i += stride) dst[i] = (__bf16)src[i];
}

// src: [D, Hc] row-major fp32 ; dst: [Hc, D] row-major bf16 (dst[h][d] = src[d][h])
__global__ void transpose_cvt(const float* __restrict__ src, __bf16* __restrict__ dst,
                              int D, int Hc) {
  int i = blockIdx.x * blockDim.x + threadIdx.x;
  int n = D * Hc;
  int stride = gridDim.x * blockDim.x;
  for (; i < n; i += stride) {
    int d = i / Hc;
    int h = i - d * Hc;
    dst[h * D + d] = (__bf16)src[i];
  }
}

// ---------------------------------------------------------------------------
// 16-bit A-matrix 16x32 fragment (contiguous k order, matching B):
// lane m = lane&15 ; element i -> K = 16*(lane>>4)+i
// ---------------------------------------------------------------------------
__device__ __forceinline__ v16bf load_a_frag_bf16(const __bf16* rowptr, int hi) {
  v8bf lo = *(const v8bf*)(rowptr + hi * 16);
  v8bf hi8 = *(const v8bf*)(rowptr + hi * 16 + 8);
  v16bf a;
#pragma unroll
  for (int i = 0; i < 8; ++i) { a[i] = lo[i]; a[8 + i] = hi8[i]; }
  return a;
}

// ---------------------------------------------------------------------------
// Projection + LayerNorm kernel.
// Block: 256 threads (8 waves). Each block: 32 rows x 512 output cols.
// Wave w owns cols [64w, 64w+64); 2 row-tiles x 4 col-tiles = 8 accumulators.
// Each B-fragment is reused by both row-tiles (halves operand traffic/WMMA).
// y is bf16; if `transposed`, store y as yT[b][col][l] (for the V matrix).
// ---------------------------------------------------------------------------
__global__ __launch_bounds__(256) void proj_ln_kernel(
    const __bf16* __restrict__ x,      // [Nrows, D] bf16
    const __bf16* __restrict__ wt,     // [512, D] bf16 (transposed weights)
    const float* __restrict__ bias,    // [512]
    const float* __restrict__ gamma,   // [512]
    const float* __restrict__ beta,    // [512]
    __bf16* __restrict__ y,
    int D, int transposed) {
  extern __shared__ float hbuf[];      // [32][512] fp32 tile = 64 KB
  __shared__ float red0[256];
  __shared__ float red1[256];
  __shared__ float rowstat[32][2];

  const int tid = threadIdx.x;
  const int wave = tid >> 5;
  const int lane = tid & 31;
  const int lo16 = lane & 15;
  const int hi = lane >> 4;
  const int r0 = blockIdx.x * 32;

  v8f acc[2][4];
#pragma unroll
  for (int h = 0; h < 2; ++h)
#pragma unroll
    for (int t = 0; t < 4; ++t)
#pragma unroll
      for (int r = 0; r < 8; ++r) acc[h][t][r] = 0.f;

  const __bf16* arow0 = x + (size_t)(r0 + lo16) * D;
  const __bf16* arow1 = x + (size_t)(r0 + 16 + lo16) * D;
  for (int kc = 0; kc < D; kc += 32) {
    if (kc + 64 < D) {
      __builtin_prefetch(arow0 + kc + 64, 0, 1);
      __builtin_prefetch(arow1 + kc + 64, 0, 1);
    }
    v16bf a0 = load_a_frag_bf16(arow0 + kc, hi);
    v16bf a1 = load_a_frag_bf16(arow1 + kc, hi);
#pragma unroll
    for (int t = 0; t < 4; ++t) {
      int col = wave * 64 + t * 16 + lo16;
      v16bf b = *(const v16bf*)(wt + (size_t)col * D + kc + hi * 16);
      acc[0][t] = WMMA_BF16(a0, b, acc[0][t]);
      acc[1][t] = WMMA_BF16(a1, b, acc[1][t]);
    }
  }

  // Spill fp32 tile (+bias) to LDS.  C layout: lane<16 -> M=r ; lane>=16 -> M=8+r
#pragma unroll
  for (int t = 0; t < 4; ++t) {
    int col = wave * 64 + t * 16 + lo16;
    float bs = bias[col];
#pragma unroll
    for (int h = 0; h < 2; ++h)
#pragma unroll
      for (int r = 0; r < 8; ++r)
        hbuf[(h * 16 + hi * 8 + r) * 512 + col] = acc[h][t][r] + bs;
  }
  __syncthreads();

  // LayerNorm over 512 cols per row: 32 rows, 8 threads/row, 64 elems/thread.
  const int row = tid >> 3;
  const int sub = tid & 7;
  float s = 0.f, q = 0.f;
#pragma unroll 8
  for (int j = 0; j < 64; ++j) {
    float v = hbuf[row * 512 + sub * 64 + j];
    s += v;
    q += v * v;
  }
  red0[tid] = s;
  red1[tid] = q;
  __syncthreads();
  if (sub == 0) {
    float S = 0.f, Q = 0.f;
#pragma unroll
    for (int j = 0; j < 8; ++j) { S += red0[row * 8 + j]; Q += red1[row * 8 + j]; }
    float mu = S * (1.f / 512.f);
    float var = Q * (1.f / 512.f) - mu * mu;
    rowstat[row][0] = mu;
    rowstat[row][1] = rsqrtf(var + 1e-5f);
  }
  __syncthreads();

  const float mu = rowstat[row][0];
  const float rstd = rowstat[row][1];
  const int grow = r0 + row;
#pragma unroll 8
  for (int j = 0; j < 64; ++j) {
    int col = sub * 64 + j;
    float v = (hbuf[row * 512 + col] - mu) * rstd * gamma[col] + beta[col];
    if (transposed) {
      int bb = grow >> 11;          // /2048
      int lr = grow & 2047;
      y[((size_t)(bb * 512 + col) << 11) + lr] = (__bf16)v;
    } else {
      y[((size_t)grow << 9) + col] = (__bf16)v;
    }
  }
}

// ---------------------------------------------------------------------------
// Attention kernel.  One block per (batch, 32 rows of l).  512 blocks total.
// Phase 1: S[32 x 2048] = K_tile · Qᵀ into 256KB LDS (bf16 WMMA, fp32 acc);
//          each Q B-fragment reused by the two K row-tiles.
// Phase 2: softmax over the 2048 axis in LDS (unnormalized exp + row sums)
// Phase 3: O = P · V via WMMA (each V B-fragment reused by two P row-tiles);
//          normalize by row sum, add residual, store fp32.
// ---------------------------------------------------------------------------
__global__ __launch_bounds__(256) void attn_kernel(
    const __bf16* __restrict__ kmat,   // [8*2048, 512] bf16 (LN'd K)
    const __bf16* __restrict__ qmat,   // [8*2048, 512] bf16 (LN'd Q)
    const __bf16* __restrict__ vT,     // [8, 512, 2048] bf16 (LN'd V, transposed)
    const float* __restrict__ resid,   // original query_sequence fp32
    float* __restrict__ out) {
  extern __shared__ float sc[];        // [32][2048] fp32 scores = 262144 B
  __shared__ float red[256];
  __shared__ float rowstat[32];

  const int tid = threadIdx.x;
  const int wave = tid >> 5;
  const int lane = tid & 31;
  const int lo16 = lane & 15;
  const int hi = lane >> 4;
  const int b = blockIdx.x >> 6;            // /64
  const int l0 = (blockIdx.x & 63) * 32;

  const __bf16* krow0 = kmat + ((size_t)(b * 2048 + l0 + lo16) << 9);
  const __bf16* krow1 = kmat + ((size_t)(b * 2048 + l0 + 16 + lo16) << 9);

  // ---- Phase 1: scores ----
  for (int chunk = 0; chunk < 2048; chunk += 128) {
    const int Lbase = chunk + wave * 16;
    const __bf16* qrow = qmat + ((size_t)(b * 2048 + Lbase + lo16) << 9);
    if (chunk + 128 < 2048)
      __builtin_prefetch(qrow + (128 << 9), 0, 1);
    v8f acc0, acc1;
#pragma unroll
    for (int r = 0; r < 8; ++r) { acc0[r] = 0.f; acc1[r] = 0.f; }
#pragma unroll 4
    for (int cb = 0; cb < 512; cb += 32) {
      v16bf bq = *(const v16bf*)(qrow + cb + hi * 16);
      v16bf a0 = load_a_frag_bf16(krow0 + cb, hi);
      v16bf a1 = load_a_frag_bf16(krow1 + cb, hi);
      acc0 = WMMA_BF16(a0, bq, acc0);
      acc1 = WMMA_BF16(a1, bq, acc1);
    }
#pragma unroll
    for (int r = 0; r < 8; ++r) {
      sc[(hi * 8 + r) * 2048 + Lbase + lo16] = acc0[r];
      sc[(16 + hi * 8 + r) * 2048 + Lbase + lo16] = acc1[r];
    }
  }
  __syncthreads();

  // ---- Phase 2: softmax (keep unnormalized; fold 1/sum into epilogue) ----
  const int row = tid >> 3;     // 32 rows, 8 threads per row
  const int sub = tid & 7;      // each thread: 256 elems
  float mx = -3.4e38f;
  for (int j = 0; j < 256; ++j) mx = fmaxf(mx, sc[row * 2048 + sub * 256 + j]);
  red[tid] = mx;
  __syncthreads();
  if (sub == 0) {
    float m = red[row * 8];
#pragma unroll
    for (int j = 1; j < 8; ++j) m = fmaxf(m, red[row * 8 + j]);
    rowstat[row] = m;
  }
  __syncthreads();
  const float rmax = rowstat[row];
  float s = 0.f;
  for (int j = 0; j < 256; ++j) {
    int idx = row * 2048 + sub * 256 + j;
    float p = __expf(sc[idx] - rmax);
    sc[idx] = p;
    s += p;
  }
  __syncthreads();   // all reads of rowstat(max) done
  red[tid] = s;
  __syncthreads();
  if (sub == 0) {
    float S = 0.f;
#pragma unroll
    for (int j = 0; j < 8; ++j) S += red[row * 8 + j];
    rowstat[row] = S;
  }
  __syncthreads();

  // ---- Phase 3: O = P · V ----
  v8f acc[2][4];
#pragma unroll
  for (int h = 0; h < 2; ++h)
#pragma unroll
    for (int t = 0; t < 4; ++t)
#pragma unroll
      for (int r = 0; r < 8; ++r) acc[h][t][r] = 0.f;

  const float* prow0 = sc + lo16 * 2048;
  const float* prow1 = sc + (16 + lo16) * 2048;
  for (int Lk = 0; Lk < 2048; Lk += 32) {
    v16bf a0, a1;
    {
      v4f f0 = *(const v4f*)(prow0 + Lk + hi * 16);
      v4f f1 = *(const v4f*)(prow0 + Lk + hi * 16 + 4);
      v4f f2 = *(const v4f*)(prow0 + Lk + hi * 16 + 8);
      v4f f3 = *(const v4f*)(prow0 + Lk + hi * 16 + 12);
#pragma unroll
      for (int i = 0; i < 4; ++i) {
        a0[i]      = (__bf16)f0[i];
        a0[4 + i]  = (__bf16)f1[i];
        a0[8 + i]  = (__bf16)f2[i];
        a0[12 + i] = (__bf16)f3[i];
      }
    }
    {
      v4f f0 = *(const v4f*)(prow1 + Lk + hi * 16);
      v4f f1 = *(const v4f*)(prow1 + Lk + hi * 16 + 4);
      v4f f2 = *(const v4f*)(prow1 + Lk + hi * 16 + 8);
      v4f f3 = *(const v4f*)(prow1 + Lk + hi * 16 + 12);
#pragma unroll
      for (int i = 0; i < 4; ++i) {
        a1[i]      = (__bf16)f0[i];
        a1[4 + i]  = (__bf16)f1[i];
        a1[8 + i]  = (__bf16)f2[i];
        a1[12 + i] = (__bf16)f3[i];
      }
    }
#pragma unroll
    for (int t = 0; t < 4; ++t) {
      int col = wave * 64 + t * 16 + lo16;
      const __bf16* vptr = vT + ((size_t)(b * 512 + col) << 11) + Lk + hi * 16;
      if (Lk + 64 < 2048) __builtin_prefetch(vptr + 64, 0, 1);
      v16bf bv = *(const v16bf*)vptr;
      acc[0][t] = WMMA_BF16(a0, bv, acc[0][t]);
      acc[1][t] = WMMA_BF16(a1, bv, acc[1][t]);
    }
  }

  // Epilogue: normalize by row sum, add residual, store fp32.
#pragma unroll
  for (int t = 0; t < 4; ++t) {
    int col = wave * 64 + t * 16 + lo16;
#pragma unroll
    for (int h = 0; h < 2; ++h)
#pragma unroll
      for (int r = 0; r < 8; ++r) {
        int m = h * 16 + hi * 8 + r;
        float inv = 1.f / rowstat[m];
        size_t idx = ((size_t)(b * 2048 + l0 + m) << 9) + col;
        out[idx] = acc[h][t][r] * inv + resid[idx];
      }
  }
}

// ---------------------------------------------------------------------------
// Launch
// ---------------------------------------------------------------------------
extern "C" void kernel_launch(void* const* d_in, const int* in_sizes, int n_in,
                              void* d_out, int out_size, void* d_ws, size_t ws_size,
                              hipStream_t stream) {
  (void)in_sizes; (void)n_in; (void)out_size; (void)ws_size;

  const float* xq_f  = (const float*)d_in[0];
  const float* xkv_f = (const float*)d_in[1];
  const float* Wq = (const float*)d_in[2];
  const float* bq = (const float*)d_in[3];
  const float* gq = (const float*)d_in[4];
  const float* betaq = (const float*)d_in[5];
  const float* Wk = (const float*)d_in[6];
  const float* bk = (const float*)d_in[7];
  const float* gk = (const float*)d_in[8];
  const float* betak = (const float*)d_in[9];
  const float* Wv = (const float*)d_in[10];
  const float* bv = (const float*)d_in[11];
  const float* gv = (const float*)d_in[12];
  const float* betav = (const float*)d_in[13];
  float* out = (float*)d_out;

  char* ws = (char*)d_ws;
  size_t off = 0;
  auto alloc = [&](size_t bytes) -> void* {
    void* p = ws + off;
    off += (bytes + 255) & ~(size_t)255;
    return p;
  };
  __bf16* xq_b  = (__bf16*)alloc((size_t)8 * 2048 * 512 * 2);
  __bf16* xkv_b = (__bf16*)alloc((size_t)8 * 2048 * 768 * 2);
  __bf16* WqT   = (__bf16*)alloc((size_t)512 * 512 * 2);
  __bf16* WkT   = (__bf16*)alloc((size_t)512 * 768 * 2);
  __bf16* WvT   = (__bf16*)alloc((size_t)512 * 768 * 2);
  __bf16* qb    = (__bf16*)alloc((size_t)8 * 2048 * 512 * 2);
  __bf16* kb    = (__bf16*)alloc((size_t)8 * 2048 * 512 * 2);
  __bf16* vTb   = (__bf16*)alloc((size_t)8 * 2048 * 512 * 2);

  cvt_f32_bf16<<<2048, 256, 0, stream>>>(xq_f, xq_b, 8 * 2048 * 512);
  cvt_f32_bf16<<<2048, 256, 0, stream>>>(xkv_f, xkv_b, 8 * 2048 * 768);
  transpose_cvt<<<512, 256, 0, stream>>>(Wq, WqT, 512, 512);
  transpose_cvt<<<768, 256, 0, stream>>>(Wk, WkT, 768, 512);
  transpose_cvt<<<768, 256, 0, stream>>>(Wv, WvT, 768, 512);

  // Q: [16384,512]x[512,512] ; K,V: [16384,768]x[768,512]; V stored transposed.
  proj_ln_kernel<<<512, 256, 32 * 512 * sizeof(float), stream>>>(
      xq_b, WqT, bq, gq, betaq, qb, 512, 0);
  proj_ln_kernel<<<512, 256, 32 * 512 * sizeof(float), stream>>>(
      xkv_b, WkT, bk, gk, betak, kb, 768, 0);
  proj_ln_kernel<<<512, 256, 32 * 512 * sizeof(float), stream>>>(
      xkv_b, WvT, bv, gv, betav, vTb, 768, 1);

  attn_kernel<<<512, 256, 32 * 2048 * sizeof(float), stream>>>(kb, qb, vTb, xq_f, out);
}
